// Transformer_67113158967550
// MI455X (gfx1250) — matrix operand order repaired and checked by reading, not compiled, not run on hardware
//
#include <hip/hip_runtime.h>
#include <hip/hip_bf16.h>

typedef __bf16 bf16_t;
typedef __attribute__((ext_vector_type(16))) __bf16 bf16x16;
typedef __attribute__((ext_vector_type(8)))  float  floatx8;
typedef unsigned int u32;

// ---------------------------------------------------------------------------
// WMMA helpers (CDNA5 / gfx1250, wave32)
// ---------------------------------------------------------------------------
__device__ __forceinline__ floatx8 wmma_bf16(bf16x16 a, bf16x16 b, floatx8 c) {
  return __builtin_amdgcn_wmma_f32_16x16x32_bf16(false, a, false, b, (short)0, c,
                                                 false, false);
}

union FragU { bf16x16 v; uint4 q[2]; };

// A-matrix 16x32 bf16 fragment (ISA 7.12.2):
//   lanes 0-15 : row M=lane,    dword pairs {0..3, 8..11}
//   lanes16-31 : row M=lane-16, dword pairs {4..7, 12..15}
// Source row-major [row, ld] bf16, K-contiguous -> two b128 loads per lane.
__device__ __forceinline__ bf16x16 load_frag_a(const bf16_t* base, long row,
                                               int ld, int k0, int lane) {
  const u32* p = (const u32*)base;
  long dw = ((row * (long)ld + k0) >> 1) + ((lane >= 16) ? 4 : 0);
  FragU u;
  u.q[0] = *(const uint4*)(p + dw);
  u.q[1] = *(const uint4*)(p + dw + 8);
  return u.v;
}

// B-matrix 32x16 bf16 fragment:
//   lanes 0-15 : col N=lane,    dword pairs {0..3, 8..11}
//   lanes16-31 : col N=lane-16, dword pairs {4..7, 12..15}
// Source row-major [n, ld] (i.e. W[n, k]), K-contiguous -> two b128 loads.
__device__ __forceinline__ bf16x16 load_frag_b(const bf16_t* base, long row,
                                               int ld, int k0, int lane) {
  const u32* p = (const u32*)base;
  long dw = ((row * (long)ld + k0) >> 1) + ((lane >= 16) ? 4 : 0);
  FragU u;
  u.q[0] = *(const uint4*)(p + dw);
  u.q[1] = *(const uint4*)(p + dw + 8);
  return u.v;
}

// ---------------------------------------------------------------------------
// fp32 -> bf16 bulk conversion
// ---------------------------------------------------------------------------
__global__ __launch_bounds__(256)
void convert_kernel(const float* __restrict__ in, bf16_t* __restrict__ out, long n) {
  long i = (long)blockIdx.x * blockDim.x + threadIdx.x;
  long stride = (long)gridDim.x * blockDim.x;
  for (; i < n; i += stride) out[i] = (bf16_t)in[i];
}

// ---------------------------------------------------------------------------
// RMSNorm (fp32 in, bf16 out), one row per block
// ---------------------------------------------------------------------------
__global__ __launch_bounds__(256)
void rmsnorm_kernel(const float* __restrict__ x, const float* __restrict__ w,
                    bf16_t* __restrict__ out, int Dm) {
  __shared__ float red[256];
  long row = blockIdx.x;
  const float* xr = x + row * (long)Dm;
  float ss = 0.f;
  for (int d = threadIdx.x; d < Dm; d += 256) { float v = xr[d]; ss += v * v; }
  red[threadIdx.x] = ss;
  __syncthreads();
  for (int s = 128; s > 0; s >>= 1) {
    if ((int)threadIdx.x < s) red[threadIdx.x] += red[threadIdx.x + s];
    __syncthreads();
  }
  float scale = rsqrtf(red[0] / (float)Dm + 1e-5f);
  for (int d = threadIdx.x; d < Dm; d += 256)
    out[row * (long)Dm + d] = (bf16_t)(xr[d] * scale * w[d]);
}

// ---------------------------------------------------------------------------
// WMMA bf16 NT-GEMM: out[M,N] = A[M,K] * W[N,K]^T.
// Double-buffered LDS pipeline: block tile 128x128, K-step 64.
// 8 waves, each computing a 32x64 sub-tile (16 v_wmma per K-step).
// Global traffic: 32KB / block / K-step for 2.1 MFLOP -> 64 flops/byte.
// Epilogues:
//   EP=0: f32 out (+ optional residual)
//   EP=1: bf16 out
//   EP=2: silu -> bf16 out
//   EP=3: f32 accumulate: out += rowscale[row*rss] * acc
// ---------------------------------------------------------------------------
template <int EP>
__global__ __launch_bounds__(256)
void gemm_nt(const bf16_t* __restrict__ A, const bf16_t* __restrict__ W,
             const float* __restrict__ residual, const float* __restrict__ rowscale,
             int rss, float* __restrict__ outf, bf16_t* __restrict__ outb,
             int M, int N, int Ktot) {
  __shared__ bf16_t sA[2][128 * 64];   // 16 KB per buffer
  __shared__ bf16_t sB[2][128 * 64];   // 16 KB per buffer (64 KB total)

  const int tid = threadIdx.x;
  const int lane = tid & 31;
  const int wave = tid >> 5;
  const int wm = wave & 3;    // 4 waves along M (32 rows each)
  const int wn = wave >> 2;   // 2 waves along N (64 cols each)
  const int m0 = blockIdx.x * 128 + wm * 32;
  const int n0 = blockIdx.y * 128 + wn * 64;

  // Staging map: thread owns half a row of each tile (16 dwords = 4 x b128).
  const int srow = tid >> 1;
  const int scol = (tid & 1) * 16;     // dword offset within 32-dword row

  int garA = blockIdx.x * 128 + srow; if (garA > M - 1) garA = M - 1;
  int garB = blockIdx.y * 128 + srow; if (garB > N - 1) garB = N - 1;
  const u32* gA = (const u32*)A + (((long)garA * Ktot) >> 1) + scol;
  const u32* gB = (const u32*)W + (((long)garB * Ktot) >> 1) + scol;

  floatx8 acc[2][4] = {};
  uint4 ra[4], rb[4];

  auto gfetch = [&](int k0) {
    const u32* pa = gA + (k0 >> 1);
    const u32* pb = gB + (k0 >> 1);
#pragma unroll
    for (int i = 0; i < 4; ++i) ra[i] = *(const uint4*)(pa + 4 * i);
#pragma unroll
    for (int i = 0; i < 4; ++i) rb[i] = *(const uint4*)(pb + 4 * i);
  };
  auto lstore = [&](int buf) {
    u32* pa = (u32*)&sA[buf][0] + srow * 32 + scol;
    u32* pb = (u32*)&sB[buf][0] + srow * 32 + scol;
#pragma unroll
    for (int i = 0; i < 4; ++i) *(uint4*)(pa + 4 * i) = ra[i];
#pragma unroll
    for (int i = 0; i < 4; ++i) *(uint4*)(pb + 4 * i) = rb[i];
  };

  const int nk = Ktot >> 6;   // K-steps of 64 (K is a multiple of 64 here)
  gfetch(0);
  lstore(0);
  int buf = 0;
  for (int kt = 0; kt < nk; ++kt) {
    __syncthreads();                          // LDS[buf] visible to all waves
    if (kt + 1 < nk) gfetch((kt + 1) << 6);   // prefetch next tile into regs
#pragma unroll
    for (int kk = 0; kk < 64; kk += 32) {
      bf16x16 a0 = load_frag_a(&sA[buf][0], wm * 32 + (lane & 15), 64, kk, lane);
      bf16x16 a1 = load_frag_a(&sA[buf][0], wm * 32 + 16 + (lane & 15), 64, kk, lane);
#pragma unroll
      for (int j = 0; j < 4; ++j) {
        bf16x16 bj = load_frag_b(&sB[buf][0], wn * 64 + j * 16 + (lane & 15), 64, kk, lane);
        acc[0][j] = wmma_bf16(a0, bj, acc[0][j]);
        acc[1][j] = wmma_bf16(a1, bj, acc[1][j]);
      }
    }
    __syncthreads();                          // all reads of LDS[buf] done
    if (kt + 1 < nk) lstore(buf ^ 1);         // flush regs to alternate buffer
    buf ^= 1;
  }

#pragma unroll
  for (int i = 0; i < 2; ++i) {
#pragma unroll
    for (int j = 0; j < 4; ++j) {
      int col = n0 + j * 16 + (lane & 15);
      if (col >= N) continue;
      int rbase = m0 + i * 16 + ((lane >= 16) ? 8 : 0);
#pragma unroll
      for (int r = 0; r < 8; ++r) {
        int row = rbase + r;
        if (row >= M) continue;
        float v = acc[i][j][r];
        long idx = (long)row * N + col;
        if (EP == 0) {
          outf[idx] = v + (residual ? residual[idx] : 0.f);
        } else if (EP == 1) {
          outb[idx] = (bf16_t)v;
        } else if (EP == 2) {
          float s = v / (1.f + __expf(-v));   // silu
          outb[idx] = (bf16_t)s;
        } else {
          outf[idx] += rowscale[(long)row * rss] * v;
        }
      }
    }
  }
}

// ---------------------------------------------------------------------------
// Flash attention: one block = one (b, h, 64-query tile). 4 waves (128 thr).
// Q/K/V/O are bf16 in [B, T, D] layout (head offset h*64, row stride Dm).
// S = (Q K^T)/8 via WMMA (K loads straight from global: natural B layout),
// online softmax in LDS, O += P V via WMMA (V staged transposed in LDS).
// ---------------------------------------------------------------------------
__global__ __launch_bounds__(128)
void flash_kernel(const bf16_t* __restrict__ Q, const bf16_t* __restrict__ Km,
                  const bf16_t* __restrict__ Vm, bf16_t* __restrict__ O,
                  int Tq, int Tk, int Dm, int Hh, int causal) {
  __shared__ float  sS[64 * 66];
  __shared__ bf16_t sP[64 * 80];
  __shared__ bf16_t sVt[64 * 80];
  __shared__ float  sAlpha[64];
  __shared__ float  sInv[64];

  const int qtiles = Tq >> 6;
  const int qt = blockIdx.x % qtiles;
  const int bh = blockIdx.x / qtiles;
  const int b = bh / Hh, h = bh % Hh;
  const int lane = threadIdx.x & 31;
  const int wave = threadIdx.x >> 5;
  const int tid = threadIdx.x;

  const bf16_t* Qb = Q + ((long)b * Tq) * Dm + h * 64;
  const bf16_t* Kb = Km + ((long)b * Tk) * Dm + h * 64;
  const bf16_t* Vb = Vm + ((long)b * Tk) * Dm + h * 64;
  bf16_t* Ob = O + ((long)b * Tq) * Dm + h * 64;

  // Persistent Q fragments (16 rows per wave, head-dim K = 0..63)
  bf16x16 qa0 = load_frag_a(Qb, qt * 64 + wave * 16 + (lane & 15), Dm, 0, lane);
  bf16x16 qa1 = load_frag_a(Qb, qt * 64 + wave * 16 + (lane & 15), Dm, 32, lane);

  floatx8 acc[4] = {};
  float m_t = -3.0e38f, l_t = 0.f;

  const int jend = causal ? qt : (Tk >> 6) - 1;
  for (int j = 0; j <= jend; ++j) {
    __syncthreads();   // previous-iteration readers done before restaging

    // ---- stage V^T into LDS: sVt[dim][key], 2 threads per key row ----
    {
      int vrow = tid >> 1;
      int d0 = (tid & 1) * 32;
      const u32* vp = (const u32*)Vb + (((long)(j * 64 + vrow) * Dm + d0) >> 1);
#pragma unroll
      for (int q4 = 0; q4 < 4; ++q4) {
        uint4 uu = *(const uint4*)(vp + q4 * 4);
        u32 vals[4] = {uu.x, uu.y, uu.z, uu.w};
#pragma unroll
        for (int e = 0; e < 4; ++e) {
          int d = d0 + q4 * 8 + e * 2;
          union { u32 u; bf16_t h2[2]; } cv; cv.u = vals[e];
          sVt[(d + 0) * 80 + vrow] = cv.h2[0];
          sVt[(d + 1) * 80 + vrow] = cv.h2[1];
        }
      }
    }

    // ---- S tiles: wave computes 16x64 strip ----
#pragma unroll
    for (int nt = 0; nt < 4; ++nt) {
      int krow = j * 64 + nt * 16 + (lane & 15);
      bf16x16 kb0 = load_frag_b(Kb, krow, Dm, 0, lane);
      bf16x16 kb1 = load_frag_b(Kb, krow, Dm, 32, lane);
      floatx8 s = {};
      s = wmma_bf16(qa0, kb0, s);
      s = wmma_bf16(qa1, kb1, s);
      int col_g = j * 64 + nt * 16 + (lane & 15);
      int rl0 = wave * 16 + ((lane >= 16) ? 8 : 0);
      int rg0 = qt * 64 + rl0;
#pragma unroll
      for (int r = 0; r < 8; ++r) {
        float v = s[r] * 0.125f;                       // 1/sqrt(DK=64)
        if (causal && col_g > rg0 + r) v = -65504.0f;  // NEG, as reference
        sS[(rl0 + r) * 66 + nt * 16 + (lane & 15)] = v;
      }
    }
    __syncthreads();

    // ---- online softmax: thread t owns query row t (t < 64) ----
    if (tid < 64) {
      float mx = -3.0e38f;
#pragma unroll 8
      for (int c = 0; c < 64; ++c) mx = fmaxf(mx, sS[tid * 66 + c]);
      float mnew = fmaxf(m_t, mx);
      float alpha = (j == 0) ? 0.f : __expf(m_t - mnew);
      float sum = 0.f;
#pragma unroll 8
      for (int c = 0; c < 64; ++c) {
        float p = __expf(sS[tid * 66 + c] - mnew);
        sum += p;
        sP[tid * 80 + c] = (bf16_t)p;
      }
      l_t = l_t * alpha + sum;
      m_t = mnew;
      sAlpha[tid] = alpha;
    }
    __syncthreads();

    // ---- rescale running O, then O += P @ V ----
    float al[8];
    {
      int rl0 = wave * 16 + ((lane >= 16) ? 8 : 0);
#pragma unroll
      for (int r = 0; r < 8; ++r) al[r] = sAlpha[rl0 + r];
    }
    bf16x16 pa0 = load_frag_a(sP, wave * 16 + (lane & 15), 80, 0, lane);
    bf16x16 pa1 = load_frag_a(sP, wave * 16 + (lane & 15), 80, 32, lane);
#pragma unroll
    for (int nt = 0; nt < 4; ++nt) {
#pragma unroll
      for (int r = 0; r < 8; ++r) acc[nt][r] *= al[r];
      bf16x16 vb0 = load_frag_b(sVt, nt * 16 + (lane & 15), 80, 0, lane);
      bf16x16 vb1 = load_frag_b(sVt, nt * 16 + (lane & 15), 80, 32, lane);
      acc[nt] = wmma_bf16(pa0, vb0, acc[nt]);
      acc[nt] = wmma_bf16(pa1, vb1, acc[nt]);
    }
  }

  __syncthreads();
  if (tid < 64) sInv[tid] = (l_t > 0.f) ? (1.f / l_t) : 0.f;
  __syncthreads();

  {
    int rl0 = wave * 16 + ((lane >= 16) ? 8 : 0);
    int rg0 = qt * 64 + rl0;
#pragma unroll
    for (int nt = 0; nt < 4; ++nt) {
      int col = nt * 16 + (lane & 15);
#pragma unroll
      for (int r = 0; r < 8; ++r) {
        float v = acc[nt][r] * sInv[rl0 + r];
        Ob[(long)(rg0 + r) * Dm + col] = (bf16_t)v;
      }
    }
  }
}

// ---------------------------------------------------------------------------
// Top-2 gating over E=8 logits (matches jax.lax.top_k tie-breaking: earliest)
// ---------------------------------------------------------------------------
__global__ __launch_bounds__(256)
void gates_kernel(const float* __restrict__ logits, float* __restrict__ gates, int n) {
  int i = blockIdx.x * blockDim.x + threadIdx.x;
  if (i >= n) return;
  float l[8];
#pragma unroll
  for (int e = 0; e < 8; ++e) l[e] = logits[(long)i * 8 + e];
  int i0 = 0;
#pragma unroll
  for (int e = 1; e < 8; ++e) if (l[e] > l[i0]) i0 = e;
  int i1 = -1;
#pragma unroll
  for (int e = 0; e < 8; ++e) {
    if (e == i0) continue;
    if (i1 < 0 || l[e] > l[i1]) i1 = e;
  }
  float e0 = 1.f;                       // exp(top - top)
  float e1 = __expf(l[i1] - l[i0]);
  float inv = 1.f / (e0 + e1);
#pragma unroll
  for (int e = 0; e < 8; ++e) gates[(long)i * 8 + e] = 0.f;
  gates[(long)i * 8 + i0] = e0 * inv;
  gates[(long)i * 8 + i1] = e1 * inv;
}

// ---------------------------------------------------------------------------
// Host orchestration
// ---------------------------------------------------------------------------
extern "C" void kernel_launch(void* const* d_in, const int* in_sizes, int n_in,
                              void* d_out, int out_size, void* d_ws, size_t ws_size,
                              hipStream_t stream) {
  (void)in_sizes; (void)n_in; (void)out_size; (void)ws_size;
  constexpr int Bq = 4, T = 1024, SE = 64, Dm = 1024, Hh = 16, E = 8, HID = 4096;
  constexpr int NTOK = Bq * T;       // 4096
  constexpr int NENC = Bq * SE;      // 256
  constexpr long DD  = (long)Dm * Dm;
  constexpr long EW  = (long)E * HID * Dm;
  constexpr long SHW = (long)HID * Dm;

  const float* x   = (const float*)d_in[0];
  const float* enc = (const float*)d_in[1];
  const float* n1w = (const float*)d_in[4];
  const float* n2w = (const float*)d_in[5];
  const float* n3w = (const float*)d_in[6];
  const float* wf[8] = {
      (const float*)d_in[7],  (const float*)d_in[8],
      (const float*)d_in[9],  (const float*)d_in[10],
      (const float*)d_in[11], (const float*)d_in[12],
      (const float*)d_in[13], (const float*)d_in[14]};
  const float* gate_w = (const float*)d_in[15];
  const float* exp_w1 = (const float*)d_in[16];
  const float* exp_w2 = (const float*)d_in[17];
  const float* sh_w1  = (const float*)d_in[18];
  const float* sh_w2  = (const float*)d_in[19];
  float* out = (float*)d_out;

  char* wp = (char*)d_ws;
  auto alloc = [&](size_t bytes) -> void* {
    void* p = (void*)wp;
    wp += (bytes + 255) & ~(size_t)255;
    return p;
  };
  bf16_t* bW[8];
  for (int i = 0; i < 8; ++i) bW[i] = (bf16_t*)alloc(DD * 2);
  bf16_t* bGate = (bf16_t*)alloc((size_t)E * Dm * 2);
  bf16_t* bEw1  = (bf16_t*)alloc(EW * 2);
  bf16_t* bEw2  = (bf16_t*)alloc(EW * 2);
  bf16_t* bSh1  = (bf16_t*)alloc(SHW * 2);
  bf16_t* bSh2  = (bf16_t*)alloc(SHW * 2);
  bf16_t* bEnc  = (bf16_t*)alloc((size_t)NENC * Dm * 2);
  bf16_t* bXn   = (bf16_t*)alloc((size_t)NTOK * Dm * 2);
  bf16_t* bQ    = (bf16_t*)alloc((size_t)NTOK * Dm * 2);
  bf16_t* bK    = (bf16_t*)alloc((size_t)NTOK * Dm * 2);
  bf16_t* bV    = (bf16_t*)alloc((size_t)NTOK * Dm * 2);
  bf16_t* bAttn = (bf16_t*)alloc((size_t)NTOK * Dm * 2);
  float*  fH    = (float*)alloc((size_t)NTOK * Dm * 4);
  float*  fH2   = (float*)alloc((size_t)NTOK * Dm * 4);
  float*  fLog  = (float*)alloc((size_t)NTOK * E * 4);
  float*  fGat  = (float*)alloc((size_t)NTOK * E * 4);
  bf16_t* bH1   = (bf16_t*)alloc((size_t)NTOK * HID * 2);

  auto conv = [&](const float* s, bf16_t* d, long n) {
    convert_kernel<<<dim3(2048), dim3(256), 0, stream>>>(s, d, n);
  };
  auto gemm = [&](int ep, const bf16_t* A, const bf16_t* Wt, const float* res,
                  const float* rsc, int rss, float* of, bf16_t* ob,
                  int M, int N, int K) {
    dim3 g((M + 127) / 128, (N + 127) / 128), blk(256);
    switch (ep) {
      case 0: gemm_nt<0><<<g, blk, 0, stream>>>(A, Wt, res, rsc, rss, of, ob, M, N, K); break;
      case 1: gemm_nt<1><<<g, blk, 0, stream>>>(A, Wt, res, rsc, rss, of, ob, M, N, K); break;
      case 2: gemm_nt<2><<<g, blk, 0, stream>>>(A, Wt, res, rsc, rss, of, ob, M, N, K); break;
      default: gemm_nt<3><<<g, blk, 0, stream>>>(A, Wt, res, rsc, rss, of, ob, M, N, K); break;
    }
  };

  // --- one-time bf16 weight conversion (stream-ordered, deterministic) ---
  for (int i = 0; i < 8; ++i) conv(wf[i], bW[i], DD);
  conv(gate_w, bGate, (long)E * Dm);
  conv(exp_w1, bEw1, EW);
  conv(exp_w2, bEw2, EW);
  conv(sh_w1, bSh1, SHW);
  conv(sh_w2, bSh2, SHW);
  conv(enc, bEnc, (long)NENC * Dm);

  // --- self-attention block: h = x + MHA(rms(x)) ---
  rmsnorm_kernel<<<dim3(NTOK), dim3(256), 0, stream>>>(x, n1w, bXn, Dm);
  gemm(1, bXn, bW[0], nullptr, nullptr, 0, nullptr, bQ, NTOK, Dm, Dm);
  gemm(1, bXn, bW[1], nullptr, nullptr, 0, nullptr, bK, NTOK, Dm, Dm);
  gemm(1, bXn, bW[2], nullptr, nullptr, 0, nullptr, bV, NTOK, Dm, Dm);
  flash_kernel<<<dim3(Bq * Hh * (T / 64)), dim3(128), 0, stream>>>(
      bQ, bK, bV, bAttn, T, T, Dm, Hh, /*causal=*/1);
  gemm(0, bAttn, bW[3], x, nullptr, 0, fH, nullptr, NTOK, Dm, Dm);

  // --- cross-attention block: h2 = h + MHA(rms(h), enc, enc) ---
  rmsnorm_kernel<<<dim3(NTOK), dim3(256), 0, stream>>>(fH, n2w, bXn, Dm);
  gemm(1, bXn, bW[4], nullptr, nullptr, 0, nullptr, bQ, NTOK, Dm, Dm);
  gemm(1, bEnc, bW[5], nullptr, nullptr, 0, nullptr, bK, NENC, Dm, Dm);
  gemm(1, bEnc, bW[6], nullptr, nullptr, 0, nullptr, bV, NENC, Dm, Dm);
  flash_kernel<<<dim3(Bq * Hh * (T / 64)), dim3(128), 0, stream>>>(
      bQ, bK, bV, bAttn, T, SE, Dm, Hh, /*causal=*/0);
  gemm(0, bAttn, bW[7], fH, nullptr, 0, fH2, nullptr, NTOK, Dm, Dm);

  // --- MoE block: out = h2 + shared + sum_e gate_e * expert_e ---
  rmsnorm_kernel<<<dim3(NTOK), dim3(256), 0, stream>>>(fH2, n3w, bXn, Dm);
  gemm(0, bXn, bGate, nullptr, nullptr, 0, fLog, nullptr, NTOK, E, Dm);
  gates_kernel<<<dim3((NTOK + 255) / 256), dim3(256), 0, stream>>>(fLog, fGat, NTOK);

  // shared expert: writes out = h2 + sh (residual folded in)
  gemm(2, bXn, bSh1, nullptr, nullptr, 0, nullptr, bH1, NTOK, HID, Dm);
  gemm(0, bH1, bSh2, fH2, nullptr, 0, out, nullptr, NTOK, Dm, HID);

  // dense experts, gated accumulation into out
  for (int e = 0; e < E; ++e) {
    gemm(2, bXn, bEw1 + (long)e * HID * Dm, nullptr, nullptr, 0, nullptr, bH1,
         NTOK, HID, Dm);
    gemm(3, bH1, bEw2 + (long)e * Dm * HID, nullptr, fGat + e, E, out, nullptr,
         NTOK, Dm, HID);
  }
}
